// TGN_541165879483
// MI455X (gfx1250) — matrix operand order repaired and checked by reading, not compiled, not run on hardware
//
#include <hip/hip_runtime.h>

// ---------------------------------------------------------------- types
typedef __bf16 bf16;
typedef __attribute__((ext_vector_type(16))) __bf16 bfvec16;
typedef __attribute__((ext_vector_type(8)))  __bf16 bfvec8;
typedef __attribute__((ext_vector_type(8)))  float  fvec8;

__device__ __forceinline__ bf16 f2bf(float f) {
    unsigned u = __float_as_uint(f);
    unsigned r = (u + 0x7FFFu + ((u >> 16) & 1u)) >> 16;
    unsigned short s = (unsigned short)r;
    return __builtin_bit_cast(bf16, s);
}
__device__ __forceinline__ float bf2f(bf16 h) {
    unsigned short s = __builtin_bit_cast(unsigned short, h);
    return __uint_as_float(((unsigned)s) << 16);
}
__device__ __forceinline__ fvec8 zero8() { fvec8 z = {}; return z; }

// Ordered-uint encoding for float atomicMax (segment max)
__device__ __forceinline__ unsigned encf(float f) {
    unsigned u = __float_as_uint(f);
    return (u & 0x80000000u) ? ~u : (u | 0x80000000u);
}
__device__ __forceinline__ float decf(unsigned u) {
    unsigned v = (u & 0x80000000u) ? (u & 0x7FFFFFFFu) : ~u;
    return __uint_as_float(v);
}

// ---------------------------------------------------------------- WMMA helpers
// A fragment: 16x32 bf16, row-major with row stride ldm (elements).
// ISA 7.12.2: lane<16 -> K groups {0..7, 16..23}; lane>=16 -> {8..15, 24..31}.
// Each group is contiguous -> two 16B LDS loads per lane.
__device__ __forceinline__ bfvec16 load_a_frag(const bf16* base, int ldm, int lane) {
    int m = lane & 15;
    int koff = (lane < 16) ? 0 : 8;
    const bf16* row = base + m * ldm + koff;
    bfvec8 lo = *(const bfvec8*)(row);
    bfvec8 hi = *(const bfvec8*)(row + 16);
    return __builtin_shufflevector(lo, hi, 0, 1, 2, 3, 4, 5, 6, 7,
                                   8, 9, 10, 11, 12, 13, 14, 15);
}
// B fragment from pre-packed weights: one aligned 32B load per lane.
// pack layout: [((kt*NT + nt)*32 + lane)*16 + e]
__device__ __forceinline__ bfvec16 load_b_pack(const bf16* __restrict__ pack,
                                               int NT, int kt, int nt, int lane) {
    return *(const bfvec16*)(pack + (((size_t)(kt * NT + nt) * 32 + lane) << 4));
}
__device__ __forceinline__ fvec8 wmma_bf16(bfvec16 a, bfvec16 b, fvec8 c) {
    return __builtin_amdgcn_wmma_f32_16x16x32_bf16(false, a, false, b, (short)0, c, false, false);
}

// ---------------------------------------------------------------- utility kernels
// Pack a KxNcols (or transposed NcolsxK) f32 weight into bf16 WMMA fragment order.
__global__ void pack_b_kernel(const float* __restrict__ src, bf16* __restrict__ dst,
                              int K, int Ncols, int transpose) {
    long long i = (long long)blockIdx.x * blockDim.x + threadIdx.x;
    long long total = (long long)K * Ncols;
    if (i >= total) return;
    int e    = (int)(i & 15);
    int lane = (int)((i >> 4) & 31);
    long long tile = i >> 9;                       // kt*NT + nt
    int NT = Ncols >> 4;
    int kt = (int)(tile / NT), nt = (int)(tile - (long long)kt * NT);
    int k = kt * 32 + ((lane < 16) ? 0 : 8) + (e & 7) + ((e >> 3) << 4);
    int n = nt * 16 + (lane & 15);
    float v = transpose ? src[(size_t)n * K + k] : src[(size_t)k * Ncols + n];
    dst[i] = f2bf(v);
}
// f32 -> bf16, 8 elements per thread (vectorized)
__global__ void cvt8_kernel(const float* __restrict__ src, bf16* __restrict__ dst,
                            long long nChunks) {
    long long i = (long long)blockIdx.x * blockDim.x + threadIdx.x;
    if (i >= nChunks) return;
    const float* s = src + i * 8;
    bfvec8 v;
#pragma unroll
    for (int j = 0; j < 8; ++j) v[j] = f2bf(s[j]);
    *(bfvec8*)(dst + i * 8) = v;
}
__global__ void win_kernel(const int* __restrict__ eDst, int* __restrict__ win, int E) {
    int e = blockIdx.x * blockDim.x + threadIdx.x;
    if (e < E) atomicMax(&win[eDst[e]], e);
}

// ---------------------------------------------------------------- edge message+GRU kernel
// One block = 32 edges (two 16-row WMMA tiles). 8 waves, each owns a 16-col tile.
// GRU gate inputs (gi/gh) stay in registers: wave w owns col tiles {w, w+8, w+16}
// of the 384-wide gi/gh, i.e. the (ir,iz,in)/(hr,hz,hn) triplet for its channels.
__launch_bounds__(256)
__global__ void edge_msg_gru_kernel(
    const bf16* __restrict__ memBf,
    const bf16* __restrict__ W1p, const float* __restrict__ b1,
    const bf16* __restrict__ W2p, const float* __restrict__ b2,
    const bf16* __restrict__ Wip, const float* __restrict__ bih,
    const bf16* __restrict__ Whp, const float* __restrict__ bhh,
    const float* __restrict__ edge_attr,
    const int* __restrict__ eSrc, const int* __restrict__ eDst,
    const int* __restrict__ win,
    bf16* __restrict__ xBf, int E)
{
    __shared__ __align__(32) bf16 sA[32][384];   // [src | dst | attr] -> [msg | dst | h1]
    __shared__ int sSrc[32], sDst[32];

    const int tid  = threadIdx.x;
    const int wave = tid >> 5;
    const int lane = tid & 31;
    const int e0   = blockIdx.x * 32;

    if (tid < 32) {
        int eg = min(e0 + tid, E - 1);
        sSrc[tid] = eSrc[eg];
        sDst[tid] = eDst[eg];
    }
    __syncthreads();

    // stage concat(src_mem, dst_mem, edge_attr) as bf16, 8-element chunks
    for (int idx = tid; idx < 32 * 48; idx += 256) {
        int i = idx / 48, cc = (idx - i * 48) * 8;
        bfvec8 v;
        if (cc < 128)      v = *(const bfvec8*)(memBf + (size_t)sSrc[i] * 128 + cc);
        else if (cc < 256) v = *(const bfvec8*)(memBf + (size_t)sDst[i] * 128 + (cc - 128));
        else {
            int eg = min(e0 + i, E - 1);
            const float* ea = edge_attr + (size_t)eg * 128 + (cc - 256);
#pragma unroll
            for (int j = 0; j < 8; ++j) v[j] = f2bf(ea[j]);
        }
        *(bfvec8*)(&sA[i][cc]) = v;
    }
    __syncthreads();

    const int n    = lane & 15;
    const int mrow = (lane < 16) ? 0 : 8;
    const int n0   = wave * 16;

    // ---- h1 = relu(in @ W1 + b1): 32x128, K=384 ----
    {
        fvec8 acc0 = zero8(), acc1 = zero8();
#pragma unroll
        for (int kt = 0; kt < 12; ++kt) {
            bfvec16 b  = load_b_pack(W1p, 8, kt, wave, lane);
            bfvec16 a0 = load_a_frag(&sA[0][kt * 32], 384, lane);
            bfvec16 a1 = load_a_frag(&sA[16][kt * 32], 384, lane);
            acc0 = wmma_bf16(a0, b, acc0);
            acc1 = wmma_bf16(a1, b, acc1);
        }
        float bias = b1[n0 + n];
        __syncthreads();                        // all GEMM1 reads of sA done
#pragma unroll
        for (int r = 0; r < 8; ++r) {
            float h0 = acc0[r] + bias, h1v = acc1[r] + bias;
            sA[mrow + r][256 + n0 + n]      = f2bf(h0 > 0.f ? h0 : 0.f);
            sA[16 + mrow + r][256 + n0 + n] = f2bf(h1v > 0.f ? h1v : 0.f);
        }
    }
    __syncthreads();

    // ---- msg = h1 @ W2 + b2 -> sA[:,0:128] ----
    {
        fvec8 acc0 = zero8(), acc1 = zero8();
#pragma unroll
        for (int kt = 0; kt < 4; ++kt) {
            bfvec16 b  = load_b_pack(W2p, 8, kt, wave, lane);
            bfvec16 a0 = load_a_frag(&sA[0][256 + kt * 32], 384, lane);
            bfvec16 a1 = load_a_frag(&sA[16][256 + kt * 32], 384, lane);
            acc0 = wmma_bf16(a0, b, acc0);
            acc1 = wmma_bf16(a1, b, acc1);
        }
        float bias = b2[n0 + n];
        __syncthreads();
#pragma unroll
        for (int r = 0; r < 8; ++r) {
            sA[mrow + r][n0 + n]      = f2bf(acc0[r] + bias);
            sA[16 + mrow + r][n0 + n] = f2bf(acc1[r] + bias);
        }
    }
    __syncthreads();

    // ---- gi = msg @ Wih^T, gh = dst @ Whh^T : 12 register accumulators ----
    fvec8 gi[2][3], gh[2][3];
#pragma unroll
    for (int rt = 0; rt < 2; ++rt)
#pragma unroll
        for (int t = 0; t < 3; ++t) { gi[rt][t] = zero8(); gh[rt][t] = zero8(); }

#pragma unroll
    for (int t = 0; t < 3; ++t) {
        int nt = wave + t * 8;                   // NT = 24
#pragma unroll
        for (int kt = 0; kt < 4; ++kt) {
            bfvec16 bi = load_b_pack(Wip, 24, kt, nt, lane);
            bfvec16 bh = load_b_pack(Whp, 24, kt, nt, lane);
#pragma unroll
            for (int rt = 0; rt < 2; ++rt) {
                bfvec16 am = load_a_frag(&sA[rt * 16][kt * 32], 384, lane);       // msg
                bfvec16 ad = load_a_frag(&sA[rt * 16][128 + kt * 32], 384, lane); // dst_mem
                gi[rt][t] = wmma_bf16(am, bi, gi[rt][t]);
                gh[rt][t] = wmma_bf16(ad, bh, gh[rt][t]);
            }
        }
    }

    // ---- GRU gates + last-write-wins scatter (all in registers) ----
    const int c = n0 + n;
    const float bi0 = bih[c], bi1 = bih[128 + c], bi2 = bih[256 + c];
    const float bh0 = bhh[c], bh1 = bhh[128 + c], bh2 = bhh[256 + c];
#pragma unroll
    for (int rt = 0; rt < 2; ++rt) {
#pragma unroll
        for (int r = 0; r < 8; ++r) {
            int m  = rt * 16 + mrow + r;
            float rg = 1.f / (1.f + __expf(-(gi[rt][0][r] + bi0 + gh[rt][0][r] + bh0)));
            float zg = 1.f / (1.f + __expf(-(gi[rt][1][r] + bi1 + gh[rt][1][r] + bh1)));
            float nn = tanhf(gi[rt][2][r] + bi2 + rg * (gh[rt][2][r] + bh2));
            float dm = bf2f(sA[m][128 + c]);
            float upd = (1.f - zg) * nn + zg * dm;
            int d = sDst[m], eg = e0 + m;
            if (eg < E && win[d] == eg)
                xBf[(size_t)d * 128 + c] = f2bf(upd);
        }
    }
}

// ---------------------------------------------------------------- node projections q/k/v/skip
__launch_bounds__(256)
__global__ void node_proj_kernel(
    const bf16* __restrict__ xBf,
    const bf16* __restrict__ Wqp, const float* __restrict__ bq,
    const bf16* __restrict__ Wkp, const float* __restrict__ bk,
    const bf16* __restrict__ Wvp, const float* __restrict__ bv,
    const bf16* __restrict__ Wsp, const float* __restrict__ bs,
    bf16* __restrict__ qBf, bf16* __restrict__ kBf, bf16* __restrict__ vBf,
    float* __restrict__ skip, int N)
{
    __shared__ __align__(32) bf16 sX[32][128];
    const int tid  = threadIdx.x;
    const int wave = tid >> 5, lane = tid & 31;
    const size_t m0 = (size_t)blockIdx.x * 32;

    for (int idx = tid; idx < 32 * 16; idx += 256) {
        int i = idx >> 4, cc = (idx & 15) * 8;
        size_t node = m0 + i; if (node >= (size_t)N) node = N - 1;
        *(bfvec8*)(&sX[i][cc]) = *(const bfvec8*)(xBf + node * 128 + cc);
    }
    __syncthreads();

    const int n    = lane & 15;
    const int mrow = (lane < 16) ? 0 : 8;

#pragma unroll
    for (int half = 0; half < 2; ++half) {
        int nt  = wave + half * 8;               // NT = 16
        int n0c = nt * 16;
        fvec8 aq[2], ak[2], av[2], as_[2];
#pragma unroll
        for (int rt = 0; rt < 2; ++rt) { aq[rt] = zero8(); ak[rt] = zero8(); av[rt] = zero8(); as_[rt] = zero8(); }
#pragma unroll
        for (int kt = 0; kt < 4; ++kt) {
            bfvec16 a0 = load_a_frag(&sX[0][kt * 32], 128, lane);
            bfvec16 a1 = load_a_frag(&sX[16][kt * 32], 128, lane);
            bfvec16 b;
            b = load_b_pack(Wqp, 16, kt, nt, lane); aq[0] = wmma_bf16(a0, b, aq[0]); aq[1] = wmma_bf16(a1, b, aq[1]);
            b = load_b_pack(Wkp, 16, kt, nt, lane); ak[0] = wmma_bf16(a0, b, ak[0]); ak[1] = wmma_bf16(a1, b, ak[1]);
            b = load_b_pack(Wvp, 16, kt, nt, lane); av[0] = wmma_bf16(a0, b, av[0]); av[1] = wmma_bf16(a1, b, av[1]);
            b = load_b_pack(Wsp, 16, kt, nt, lane); as_[0] = wmma_bf16(a0, b, as_[0]); as_[1] = wmma_bf16(a1, b, as_[1]);
        }
        float bq_ = bq[n0c + n], bk_ = bk[n0c + n], bv_ = bv[n0c + n], bs_ = bs[n0c + n];
#pragma unroll
        for (int rt = 0; rt < 2; ++rt) {
#pragma unroll
            for (int r = 0; r < 8; ++r) {
                size_t node = m0 + rt * 16 + mrow + r;
                if (node < (size_t)N) {
                    size_t o = node * 256 + n0c + n;
                    qBf[o]  = f2bf(aq[rt][r] + bq_);
                    kBf[o]  = f2bf(ak[rt][r] + bk_);
                    vBf[o]  = f2bf(av[rt][r] + bv_);
                    skip[o] = as_[rt][r] + bs_;
                }
            }
        }
    }
}

// ---------------------------------------------------------------- attention
__global__ void attn_logits_kernel(const bf16* __restrict__ qBf, const bf16* __restrict__ kBf,
                                   const int* __restrict__ eSrc, const int* __restrict__ eDst,
                                   float* __restrict__ logits, unsigned* __restrict__ mEnc, int E)
{
    int idx = blockIdx.x * blockDim.x + threadIdx.x;
    if (idx >= E * 2) return;
    int e = idx >> 1, h = idx & 1;
    int s = eSrc[e], d = eDst[e];
    const bfvec16* qv = (const bfvec16*)(qBf + (size_t)d * 256 + h * 128);
    const bfvec16* kv = (const bfvec16*)(kBf + (size_t)s * 256 + h * 128);
    float acc = 0.f;
#pragma unroll
    for (int j = 0; j < 8; ++j) {
        bfvec16 qx = qv[j], kx = kv[j];
#pragma unroll
        for (int c = 0; c < 16; ++c) acc += bf2f(qx[c]) * bf2f(kx[c]);
    }
    float l = acc * 0.08838834764831845f;   // 1/sqrt(128)
    logits[idx] = l;
    atomicMax(&mEnc[(size_t)d * 2 + h], encf(l));
}

__global__ void attn_exp_kernel(const float* __restrict__ logits, const unsigned* __restrict__ mEnc,
                                const int* __restrict__ eDst, float* __restrict__ ee,
                                float* __restrict__ sSum, int E)
{
    int idx = blockIdx.x * blockDim.x + threadIdx.x;
    if (idx >= E * 2) return;
    int e = idx >> 1, h = idx & 1;
    int d = eDst[e];
    float v = __expf(logits[idx] - decf(mEnc[(size_t)d * 2 + h]));
    ee[idx] = v;
    atomicAdd(&sSum[(size_t)d * 2 + h], v);
}

__global__ void attn_agg_kernel(const float* __restrict__ ee, const float* __restrict__ sSum,
                                const bf16* __restrict__ vBf,
                                const int* __restrict__ eSrc, const int* __restrict__ eDst,
                                float* __restrict__ agg, long long total)
{
    long long idx = (long long)blockIdx.x * blockDim.x + threadIdx.x;
    if (idx >= total) return;
    int e  = (int)(idx >> 8);
    int hc = (int)(idx & 255);
    int h  = hc >> 7;
    int s = eSrc[e], d = eDst[e];
    float alpha = ee[(size_t)e * 2 + h] / sSum[(size_t)d * 2 + h];
    atomicAdd(&agg[(size_t)d * 256 + hc], alpha * bf2f(vBf[(size_t)s * 256 + hc]));
}

// ---------------------------------------------------------------- classifier
__launch_bounds__(128)
__global__ void out_kernel(const float* __restrict__ agg, const float* __restrict__ skip,
                           const bf16* __restrict__ Wcp, const float* __restrict__ bc,
                           float* __restrict__ out, int N)
{
    __shared__ __align__(32) bf16 sO[32][256];
    const int tid  = threadIdx.x;
    const int wave = tid >> 5, lane = tid & 31;
    const size_t m0 = (size_t)blockIdx.x * 32;

    for (int idx = tid; idx < 32 * 32; idx += 128) {
        int i = idx >> 5, cc = (idx & 31) * 8;
        size_t node = m0 + i; if (node >= (size_t)N) node = N - 1;
        const float* a = agg  + node * 256 + cc;
        const float* s = skip + node * 256 + cc;
        bfvec8 v;
#pragma unroll
        for (int j = 0; j < 8; ++j) v[j] = f2bf(a[j] + s[j]);
        *(bfvec8*)(&sO[i][cc]) = v;
    }
    __syncthreads();

    const int n    = lane & 15;
    const int mrow = (lane < 16) ? 0 : 8;
    const int nt   = wave;                       // NT = 4
    fvec8 acc0 = zero8(), acc1 = zero8();
#pragma unroll
    for (int kt = 0; kt < 8; ++kt) {
        bfvec16 b  = load_b_pack(Wcp, 4, kt, nt, lane);
        bfvec16 a0 = load_a_frag(&sO[0][kt * 32], 256, lane);
        bfvec16 a1 = load_a_frag(&sO[16][kt * 32], 256, lane);
        acc0 = wmma_bf16(a0, b, acc0);
        acc1 = wmma_bf16(a1, b, acc1);
    }
    float bias = bc[nt * 16 + n];
#pragma unroll
    for (int r = 0; r < 8; ++r) {
        size_t node0 = m0 + mrow + r;
        size_t node1 = m0 + 16 + mrow + r;
        if (node0 < (size_t)N) out[node0 * 64 + nt * 16 + n] = acc0[r] + bias;
        if (node1 < (size_t)N) out[node1 * 64 + nt * 16 + n] = acc1[r] + bias;
    }
}

// ---------------------------------------------------------------- host side
extern "C" void kernel_launch(void* const* d_in, const int* in_sizes, int n_in,
                              void* d_out, int out_size, void* d_ws, size_t ws_size,
                              hipStream_t stream) {
    (void)n_in; (void)out_size; (void)ws_size;
    const int N = in_sizes[0] / 128;       // memory [N,128]
    const int E = in_sizes[21] / 2;        // edge_index [2,E]

    const float* memory = (const float*)d_in[0];
    const float* W1  = (const float*)d_in[1];
    const float* b1  = (const float*)d_in[2];
    const float* W2  = (const float*)d_in[3];
    const float* b2  = (const float*)d_in[4];
    const float* Wih = (const float*)d_in[5];
    const float* Whh = (const float*)d_in[6];
    const float* bih = (const float*)d_in[7];
    const float* bhh = (const float*)d_in[8];
    const float* Wq  = (const float*)d_in[9];
    const float* bq  = (const float*)d_in[10];
    const float* Wk  = (const float*)d_in[11];
    const float* bk  = (const float*)d_in[12];
    const float* Wv  = (const float*)d_in[13];
    const float* bv  = (const float*)d_in[14];
    const float* Ws  = (const float*)d_in[15];
    const float* bs  = (const float*)d_in[16];
    const float* Wc  = (const float*)d_in[17];
    const float* bc  = (const float*)d_in[18];
    const float* edge_attr = (const float*)d_in[19];
    // d_in[20] = edge_time (unused by the reference)
    const int* edge_index = (const int*)d_in[21];
    const int* eSrc = edge_index;
    const int* eDst = edge_index + E;
    float* out = (float*)d_out;

    // ---- workspace carve ----
    char* p = (char*)d_ws;
    auto alloc = [&](size_t bytes) -> void* {
        void* r = (void*)p;
        p += (bytes + 255) & ~(size_t)255;
        return r;
    };
    bf16* W1p   = (bf16*)alloc((size_t)384 * 128 * 2);
    bf16* W2p   = (bf16*)alloc((size_t)128 * 128 * 2);
    bf16* Wip   = (bf16*)alloc((size_t)128 * 384 * 2);
    bf16* Whp   = (bf16*)alloc((size_t)128 * 384 * 2);
    bf16* Wqp   = (bf16*)alloc((size_t)128 * 256 * 2);
    bf16* Wkp   = (bf16*)alloc((size_t)128 * 256 * 2);
    bf16* Wvp   = (bf16*)alloc((size_t)128 * 256 * 2);
    bf16* Wsp   = (bf16*)alloc((size_t)128 * 256 * 2);
    bf16* Wcp   = (bf16*)alloc((size_t)256 * 64 * 2);
    bf16* memBf = (bf16*)alloc((size_t)N * 128 * 2);
    bf16* xBf   = (bf16*)alloc((size_t)N * 128 * 2);
    bf16* qBf   = (bf16*)alloc((size_t)N * 256 * 2);
    bf16* kBf   = (bf16*)alloc((size_t)N * 256 * 2);
    bf16* vBf   = (bf16*)alloc((size_t)N * 256 * 2);
    float* skip = (float*)alloc((size_t)N * 256 * 4);
    float* agg  = (float*)alloc((size_t)N * 256 * 4);
    int* win        = (int*)alloc((size_t)N * 4);
    unsigned* mEnc  = (unsigned*)alloc((size_t)N * 2 * 4);
    float* sSum     = (float*)alloc((size_t)N * 2 * 4);
    float* logits   = (float*)alloc((size_t)E * 2 * 4);
    float* ee       = (float*)alloc((size_t)E * 2 * 4);

    auto pack = [&](const float* s, bf16* d, int K, int Ncols, int tr) {
        long long total = (long long)K * Ncols;
        pack_b_kernel<<<(unsigned)((total + 255) / 256), 256, 0, stream>>>(s, d, K, Ncols, tr);
    };

    // 1) pack weights into WMMA fragment order + convert memory
    pack(W1, W1p, 384, 128, 0);
    pack(W2, W2p, 128, 128, 0);
    pack(Wih, Wip, 128, 384, 1);   // B = Wih^T
    pack(Whh, Whp, 128, 384, 1);   // B = Whh^T
    pack(Wq, Wqp, 128, 256, 0);
    pack(Wk, Wkp, 128, 256, 0);
    pack(Wv, Wvp, 128, 256, 0);
    pack(Ws, Wsp, 128, 256, 0);
    pack(Wc, Wcp, 256, 64, 0);
    {
        long long chunks = (long long)N * 128 / 8;
        cvt8_kernel<<<(unsigned)((chunks + 255) / 256), 256, 0, stream>>>(memory, memBf, chunks);
    }
    hipMemcpyAsync(xBf, memBf, (size_t)N * 128 * 2, hipMemcpyDeviceToDevice, stream);

    // 2) flag/zero init
    hipMemsetAsync(win, 0xFF, (size_t)N * 4, stream);                 // -1
    hipMemsetAsync(mEnc, 0x00, (size_t)N * 2 * 4, stream);            // -inf floor
    hipMemsetAsync(sSum, 0x00, (size_t)N * 2 * 4, stream);
    hipMemsetAsync(agg,  0x00, (size_t)N * 256 * 4, stream);

    // 3) last-write-wins winner map
    win_kernel<<<(E + 255) / 256, 256, 0, stream>>>(eDst, win, E);

    // 4) edge message MLP + GRU + scatter   (32 edges / block)
    edge_msg_gru_kernel<<<(E + 31) / 32, 256, 0, stream>>>(
        memBf, W1p, b1, W2p, b2, Wip, bih, Whp, bhh,
        edge_attr, eSrc, eDst, win, xBf, E);

    // 5) node projections q/k/v + skip    (32 nodes / block)
    node_proj_kernel<<<(N + 31) / 32, 256, 0, stream>>>(
        xBf, Wqp, bq, Wkp, bk, Wvp, bv, Wsp, bs,
        qBf, kBf, vBf, skip, N);

    // 6) attention: logits + segment max, exp + segment sum, weighted aggregate
    attn_logits_kernel<<<(2 * E + 255) / 256, 256, 0, stream>>>(qBf, kBf, eSrc, eDst, logits, mEnc, E);
    attn_exp_kernel<<<(2 * E + 255) / 256, 256, 0, stream>>>(logits, mEnc, eDst, ee, sSum, E);
    long long total = (long long)E * 256;
    attn_agg_kernel<<<(unsigned)((total + 255) / 256), 256, 0, stream>>>(ee, sSum, vBf, eSrc, eDst, agg, total);

    // 7) classifier   (32 nodes / block)
    out_kernel<<<(N + 31) / 32, 128, 0, stream>>>(agg, skip, Wcp, bc, out, N);
}
